// GBA_LBA_4930622456022
// MI455X (gfx1250) — compile-verified, matrix operand order checked by reading, not run on hardware
//
#include <hip/hip_runtime.h>

typedef unsigned short u16;
typedef __attribute__((ext_vector_type(16))) __bf16 v16bf;
typedef __attribute__((ext_vector_type(8)))  float  v8f;

// ---------- helpers ----------
__device__ __forceinline__ u16 f2bf(float f) {
    unsigned int u = __float_as_uint(f);
    unsigned int r = (u + 0x7FFFu + ((u >> 16) & 1u)) >> 16;
    return (u16)r;
}

__device__ __forceinline__ v8f wmma_bf16(v16bf a, v16bf b, v8f c) {
    return __builtin_amdgcn_wmma_f32_16x16x32_bf16(false, a, false, b, (short)0, c, false, false);
}

// ---------- pack kernels ----------
__global__ void pack_f32_bf16(const float* __restrict__ src, u16* __restrict__ dst, int n) {
    int i = blockIdx.x * 256 + threadIdx.x;
    if (i < n) dst[i] = f2bf(src[i]);
}

// x (b,128,96,96) f32 CHW  ->  xt (b,9216,128) bf16 HWC
__global__ void pack_x_hwc(const float* __restrict__ x, u16* __restrict__ xt) {
    int i = blockIdx.x * 256 + threadIdx.x;      // b*9216*128 threads
    int c = i & 127;
    int n = (i >> 7) % 9216;
    int b = i / (9216 * 128);
    xt[i] = f2bf(x[((size_t)b * 128 + c) * 9216 + n]);
}

// avgpool 4x4 of x -> xp_t (b,576,128) bf16 HWC
__global__ void avgpool_pack(const float* __restrict__ x, u16* __restrict__ xp_t) {
    int i = blockIdx.x * 256 + threadIdx.x;      // 8*576*128 threads
    int c = i & 127;
    int kp = (i >> 7) % 576;
    int b = i / (576 * 128);
    int py = kp / 24, px = kp % 24;
    const float* base = x + ((size_t)b * 128 + c) * 9216 + (py * 4) * 96 + px * 4;
    float s = 0.f;
#pragma unroll
    for (int dy = 0; dy < 4; ++dy)
#pragma unroll
        for (int dx = 0; dx < 4; ++dx) s += base[dy * 96 + dx];
    xp_t[i] = f2bf(s * 0.0625f);
}

// ---------- generic WMMA GEMM  D = A(MxK) * Bt(NxK)^T + bias ----------
// B (activation) fragments are hoisted into registers once per wave; the wave
// then loops over all MT m-tiles so each activation byte is read from HBM once.
// EPI 0: f32 CHW store             (qkv, proj)
// EPI 1: bf16 transposed store     (gq -> qlo_t)
// EPI 2: SiLU + bf16 transposed    (am1 -> a1_t)
// EPI 3: tanh(0.25*a)*v + bf16 T   (am2 -> cat_t cols 0..63)
// EPI 4: gkv split: kk -> transposed (NxK), vv -> d-major
template<int K, int EPI, int MT>
__global__ __launch_bounds__(128) void gemm_wmma(
    const u16* __restrict__ A, const u16* __restrict__ Bt,
    const float* __restrict__ bias, int N,
    float* __restrict__ outF, u16* __restrict__ outT, u16* __restrict__ outT2,
    int ldt, int col_off, const float* __restrict__ vbuf)
{
    constexpr int KS = K / 32;
    constexpr int M  = MT * 16;
    const int lane = threadIdx.x & 31;
    const int wave = threadIdx.x >> 5;
    const int r15  = lane & 15;
    const int half = lane >> 4;
    const int n0 = blockIdx.x * 64 + wave * 16;
    const int b  = blockIdx.z;

    // hoist activation (B) fragments for this wave's 16 columns
    v16bf bfrag[KS];
    {
        const u16* brow = Bt + ((size_t)b * N + n0 + r15) * K + half * 8;
#pragma unroll
        for (int ks = 0; ks < KS; ++ks) {
            uint4* bv = (uint4*)&bfrag[ks];
            bv[0] = *(const uint4*)(brow + ks * 32);
            bv[1] = *(const uint4*)(brow + ks * 32 + 16);
        }
    }

#pragma unroll
    for (int mt = 0; mt < MT; ++mt) {
        const int m0 = mt * 16;
        const u16* arow = A + (size_t)(m0 + r15) * K + half * 8;
        v8f acc = {};
#pragma unroll
        for (int ks = 0; ks < KS; ++ks) {
            v16bf af;
            uint4* av = (uint4*)&af;
            av[0] = *(const uint4*)(arow + ks * 32);
            av[1] = *(const uint4*)(arow + ks * 32 + 16);
            acc = wmma_bf16(af, bfrag[ks], acc);
        }

        if (EPI == 0) {
            float* out = outF + ((size_t)b * M + m0) * N;
#pragma unroll
            for (int r = 0; r < 8; ++r) {
                int m = r + 8 * half;
                out[(size_t)m * N + n0 + r15] = acc[r] + bias[m0 + m];
            }
        } else if (EPI == 4) {
            if (m0 < 64) {  // kk half: transposed NxK bf16
                alignas(16) u16 h8[8];
#pragma unroll
                for (int r = 0; r < 8; ++r)
                    h8[r] = f2bf(acc[r] + bias[m0 + r + 8 * half]);
                u16* dst = outT + ((size_t)b * N + n0 + r15) * 64 + m0 + half * 8;
                *(uint4*)dst = *(const uint4*)h8;
            } else {        // vv half: d-major [b][64][576] bf16
#pragma unroll
                for (int r = 0; r < 8; ++r) {
                    int m = m0 + r + 8 * half;
                    outT2[((size_t)b * 64 + (m - 64)) * 576 + n0 + r15] =
                        f2bf(acc[r] + bias[m]);
                }
            }
        } else {
            alignas(16) u16 h8[8];
#pragma unroll
            for (int r = 0; r < 8; ++r) {
                int m = m0 + r + 8 * half;
                float a0 = acc[r] + bias[m];
                float val;
                if (EPI == 1) {
                    val = a0;
                } else if (EPI == 2) {
                    val = a0 / (1.0f + __expf(-a0));        // SiLU
                } else {                                    // EPI 3
                    float t = tanhf(a0 * 0.25f);
                    val = t * vbuf[((size_t)b * M + m) * N + n0 + r15];
                }
                h8[r] = f2bf(val);
            }
            u16* dst = outT + ((size_t)b * N + n0 + r15) * ldt + col_off + m0 + half * 8;
            *(uint4*)dst = *(const uint4*)h8;
        }
    }
}

// ---------- depthwise 5x5 on qkv_lin; emit q*k (bf16 NxC) and v (f32 CHW) ----------
__global__ __launch_bounds__(256) void dwconv_qk_v(
    const float* __restrict__ qkv_lin, const float* __restrict__ dw_w,
    const float* __restrict__ dw_b, u16* __restrict__ qk_t, float* __restrict__ v_buf)
{
    __shared__ float sm[3][20][20];
    const int tile = blockIdx.x;             // 36 spatial tiles
    const int ch   = blockIdx.y;             // 64
    const int b    = blockIdx.z;             // 8
    const int tx0 = (tile % 6) * 16, ty0 = (tile / 6) * 16;
    const int tx = threadIdx.x, ty = threadIdx.y;
    const int tid = ty * 16 + tx;
    const size_t plane = 9216;
    const float* base = qkv_lin + (size_t)b * 192 * plane;
#pragma unroll
    for (int p = 0; p < 3; ++p) {
        const float* src = base + (size_t)(p * 64 + ch) * plane;
        for (int l = tid; l < 400; l += 256) {
            int ly = l / 20, lx = l % 20;
            int gy = ty0 + ly - 2, gx = tx0 + lx - 2;
            float v = 0.f;
            if (gy >= 0 && gy < 96 && gx >= 0 && gx < 96) v = src[gy * 96 + gx];
            sm[p][ly][lx] = v;
        }
    }
    __syncthreads();
    const float* wq = dw_w + (size_t)ch * 25;
    const float* wk = dw_w + (size_t)(64 + ch) * 25;
    const float* wv = dw_w + (size_t)(128 + ch) * 25;
    float aq = 0.f, ak = 0.f, avv = 0.f;
#pragma unroll
    for (int ky = 0; ky < 5; ++ky)
#pragma unroll
        for (int kx = 0; kx < 5; ++kx) {
            int o = ky * 5 + kx;
            aq  += wq[o] * sm[0][ty + ky][tx + kx];
            ak  += wk[o] * sm[1][ty + ky][tx + kx];
            avv += wv[o] * sm[2][ty + ky][tx + kx];
        }
    float qv = aq + dw_b[ch];
    float kv = ak + dw_b[64 + ch];
    float vv = avv + dw_b[128 + ch];
    int n = (ty0 + ty) * 96 + tx0 + tx;
    qk_t[((size_t)b * 9216 + n) * 64 + ch] = f2bf(qv * kv);
    v_buf[((size_t)b * 64 + ch) * 9216 + n] = vv;
}

// ---------- grouped attention: one wave per 16 query rows ----------
__global__ __launch_bounds__(32) void attn_kernel(
    const u16* __restrict__ qlo_t,   // [b][9216][64]  bf16 (head-major cols)
    const u16* __restrict__ kkt,     // [b][576][64]   bf16
    const u16* __restrict__ vvT,     // [b][64][576]   bf16 (d-major)
    u16* __restrict__ cat_t)         // [b][9216][128] bf16, write cols 64..127
{
    __shared__ float sc[16 * 576];
    __shared__ u16   pr[16 * 576];
    __shared__ float rden[16];

    const int it = blockIdx.x;       // 576 query tiles
    const int h  = blockIdx.y;       // 4 heads
    const int b  = blockIdx.z;       // 8
    const int lane = threadIdx.x;
    const int r15 = lane & 15, half = lane >> 4;
    const uint4 z4 = make_uint4(0u, 0u, 0u, 0u);

    // prefetch this lane's V row so it streams in under scores+softmax
    const u16* vrow = vvT + ((size_t)b * 64 + h * 16 + r15) * 576;
#pragma unroll
    for (int o = 0; o < 576; o += 64)
        __builtin_prefetch(vrow + o, 0, 1);

    // Q fragment (K=16 padded to 32 with zeros)
    v16bf aq; uint4* aqv = (uint4*)&aq;
    const u16* qrow = qlo_t + ((size_t)b * 9216 + it * 16 + r15) * 64 + h * 16;
    aqv[0] = *(const uint4*)(qrow + half * 8);
    aqv[1] = z4;

    // scores = 0.25 * Q K^T  -> LDS
    const u16* kbase = kkt + (size_t)b * 576 * 64 + h * 16;
    for (int jt = 0; jt < 36; ++jt) {
        v16bf bk; uint4* bkv = (uint4*)&bk;
        const u16* krow = kbase + (size_t)(jt * 16 + r15) * 64;
        bkv[0] = *(const uint4*)(krow + half * 8);
        bkv[1] = z4;
        v8f s = {};
        s = wmma_bf16(aq, bk, s);
#pragma unroll
        for (int r = 0; r < 8; ++r)
            sc[(r + 8 * half) * 576 + jt * 16 + r15] = s[r] * 0.25f;
    }
    __syncthreads();

    // softmax over 576: 2 lanes per row
    {
        const int row = r15;
        float mx = -1e30f;
        for (int j = half * 288; j < half * 288 + 288; ++j)
            mx = fmaxf(mx, sc[row * 576 + j]);
        mx = fmaxf(mx, __shfl_xor(mx, 16, 32));
        float sum = 0.f;
        for (int j = half * 288; j < half * 288 + 288; ++j) {
            float p = __expf(sc[row * 576 + j] - mx);
            pr[row * 576 + j] = f2bf(p);
            sum += p;
        }
        sum += __shfl_xor(sum, 16, 32);
        if (half == 0) rden[row] = 1.0f / sum;
    }
    __syncthreads();

    // lo = P @ V   (K = 576)
    v8f acc = {};
    const u16* prow = pr + (size_t)r15 * 576;
    for (int kt = 0; kt < 18; ++kt) {
        v16bf ap, bv;
        uint4* apv = (uint4*)&ap;
        uint4* bvv = (uint4*)&bv;
        int k0 = kt * 32;
        apv[0] = *(const uint4*)(prow + k0 + half * 8);
        apv[1] = *(const uint4*)(prow + k0 + 16 + half * 8);
        bvv[0] = *(const uint4*)(vrow + k0 + half * 8);
        bvv[1] = *(const uint4*)(vrow + k0 + 16 + half * 8);
        acc = wmma_bf16(ap, bv, acc);
    }
#pragma unroll
    for (int r = 0; r < 8; ++r) {
        int il = r + 8 * half;                 // local query row
        int i  = it * 16 + il;                 // global spatial index
        float val = acc[r] * rden[il];
        cat_t[((size_t)b * 9216 + i) * 128 + 64 + h * 16 + r15] = f2bf(val);
    }
}

// ---------- host ----------
extern "C" void kernel_launch(void* const* d_in, const int* in_sizes, int n_in,
                              void* d_out, int out_size, void* d_ws, size_t ws_size,
                              hipStream_t stream)
{
    (void)in_sizes; (void)n_in; (void)out_size; (void)ws_size;
    const float* x      = (const float*)d_in[0];
    const float* qkv_w  = (const float*)d_in[1];
    const float* qkv_b  = (const float*)d_in[2];
    const float* dw_w   = (const float*)d_in[3];
    const float* dw_b   = (const float*)d_in[4];
    const float* am_w1  = (const float*)d_in[5];
    const float* am_b1  = (const float*)d_in[6];
    const float* am_w2  = (const float*)d_in[7];
    const float* am_b2  = (const float*)d_in[8];
    const float* gq_w   = (const float*)d_in[9];
    const float* gq_b   = (const float*)d_in[10];
    const float* gkv_w  = (const float*)d_in[11];
    const float* gkv_b  = (const float*)d_in[12];
    const float* proj_w = (const float*)d_in[13];
    const float* proj_b = (const float*)d_in[14];
    float* out = (float*)d_out;

    char* ws = (char*)d_ws;
    size_t off = 0;
    auto carve = [&](size_t bytes) -> void* {
        void* p = ws + off;
        off += (bytes + 255) & ~(size_t)255;
        return p;
    };
    u16*   xt    = (u16*)carve((size_t)8 * 9216 * 128 * 2);   // x HWC bf16
    float* qkvl  = (float*)carve((size_t)8 * 192 * 9216 * 4); // qkv conv1x1 out
    u16*   qk_t  = (u16*)carve((size_t)8 * 9216 * 64 * 2);    // q*k  (NxC)
    float* v_buf = (float*)carve((size_t)8 * 64 * 9216 * 4);  // v    (CHW)
    u16*   a1_t  = (u16*)carve((size_t)8 * 9216 * 64 * 2);    // silu(am1)
    u16*   qlo_t = (u16*)carve((size_t)8 * 9216 * 64 * 2);    // low-branch q
    u16*   cat_t = (u16*)carve((size_t)8 * 9216 * 128 * 2);   // concat(hi,lo) NxC
    u16*   xp_t  = (u16*)carve((size_t)8 * 576 * 128 * 2);    // pooled x HWC
    u16*   kkt   = (u16*)carve((size_t)8 * 576 * 64 * 2);     // kk NxK
    u16*   vvT   = (u16*)carve((size_t)8 * 64 * 576 * 2);     // vv d-major
    u16*   qkvwb = (u16*)carve((size_t)192 * 128 * 2);
    u16*   am1b  = (u16*)carve((size_t)64 * 64 * 2);
    u16*   am2b  = (u16*)carve((size_t)64 * 64 * 2);
    u16*   gqb   = (u16*)carve((size_t)64 * 128 * 2);
    u16*   gkvb  = (u16*)carve((size_t)128 * 128 * 2);
    u16*   projb = (u16*)carve((size_t)128 * 128 * 2);

    // pack weights + activations to bf16
    auto packw = [&](const float* s, u16* d, int n) {
        pack_f32_bf16<<<dim3((n + 255) / 256), dim3(256), 0, stream>>>(s, d, n);
    };
    packw(qkv_w, qkvwb, 192 * 128);
    packw(am_w1, am1b, 64 * 64);
    packw(am_w2, am2b, 64 * 64);
    packw(gq_w,  gqb,  64 * 128);
    packw(gkv_w, gkvb, 128 * 128);
    packw(proj_w, projb, 128 * 128);
    pack_x_hwc<<<dim3(8 * 9216 * 128 / 256), dim3(256), 0, stream>>>(x, xt);

    // qkv conv1x1 (M=192,K=128,N=9216) -> f32 CHW ; activations read once
    gemm_wmma<128, 0, 12><<<dim3(144, 1, 8), dim3(128), 0, stream>>>(
        qkvwb, xt, qkv_b, 9216, qkvl, nullptr, nullptr, 0, 0, nullptr);

    // depthwise 5x5 + q*k / v
    dwconv_qk_v<<<dim3(36, 64, 8), dim3(16, 16), 0, stream>>>(
        qkvl, dw_w, dw_b, qk_t, v_buf);

    // am1 + SiLU (M=64,K=64)
    gemm_wmma<64, 2, 4><<<dim3(144, 1, 8), dim3(128), 0, stream>>>(
        am1b, qk_t, am_b1, 9216, nullptr, a1_t, nullptr, 64, 0, nullptr);

    // am2 -> tanh(0.25*a)*v -> hi half of cat_t
    gemm_wmma<64, 3, 4><<<dim3(144, 1, 8), dim3(128), 0, stream>>>(
        am2b, a1_t, am_b2, 9216, nullptr, cat_t, nullptr, 128, 0, v_buf);

    // gq conv1x1 (M=64,K=128) -> qlo_t
    gemm_wmma<128, 1, 4><<<dim3(144, 1, 8), dim3(128), 0, stream>>>(
        gqb, xt, gq_b, 9216, nullptr, qlo_t, nullptr, 64, 0, nullptr);

    // avgpool + pack
    avgpool_pack<<<dim3(8 * 576 * 128 / 256), dim3(256), 0, stream>>>(x, xp_t);

    // gkv conv1x1 (M=128,K=128,N=576) -> kk / vv
    gemm_wmma<128, 4, 8><<<dim3(9, 1, 8), dim3(128), 0, stream>>>(
        gkvb, xp_t, gkv_b, 576, nullptr, kkt, vvT, 64, 0, nullptr);

    // grouped attention -> lo half of cat_t
    attn_kernel<<<dim3(576, 4, 8), dim3(32), 0, stream>>>(qlo_t, kkt, vvT, cat_t);

    // final projection (M=128,K=128) -> f32 output
    gemm_wmma<128, 0, 8><<<dim3(144, 1, 8), dim3(128), 0, stream>>>(
        projb, cat_t, proj_b, 9216, out, nullptr, nullptr, 0, 0, nullptr);
}
